// GDL_52982716564225
// MI455X (gfx1250) — matrix operand order repaired and checked by reading, not compiled, not run on hardware
//
#include <hip/hip_runtime.h>

typedef __attribute__((ext_vector_type(16))) __bf16 v16bf;
typedef __attribute__((ext_vector_type(8)))  __bf16 v8bf;
typedef __attribute__((ext_vector_type(8)))  float  v8f;
typedef long long ll;

constexpr int B_   = 8;
constexpr int N_   = 2048;
constexpr int E_   = 32768;
constexpr int EN_  = E_ + N_;      // edges + self loops
constexpr int FIN_ = 512;
constexpr int F1_  = 256;
constexpr int F2_  = 128;
constexpr int K1_  = 100;
constexpr int K2_  = 50;
constexpr int D_   = 128;
constexpr int S_   = 51;           // 1 cls + 50 pooled tokens

// ---------------------------------------------------------------------------
// Generic WMMA GEMM:  C[M,N] = act(A[M,K] * B[K,N] + bias)
// ABF=0: A is f32 row-major, converted to bf16 in registers.
// ABF=1: A is bf16 row-major [M][K] loaded directly.
// Bt: bf16 stored [N][K] so each lane's fragment is one aligned 32B read.
// One wave computes a 16x64 C tile (4 accumulators share one A fragment per
// 32-deep k-step).  Out-of-range rows/cols are CLAMPED on load (not
// predicated): a garbage A-row m only corrupts C-row m, a garbage B-col n
// only corrupts C-col n, neither of which is stored -> branch-free k-loop.
// A and B fragments both use contiguous k = 16*(lane/16) + s (wave32).
// ---------------------------------------------------------------------------
static __device__ __forceinline__ void cvt8(v16bf& af, int base, const float* p) {
    float4 a = *(const float4*)p;
    float4 b = *(const float4*)(p + 4);
    af[base + 0] = (__bf16)a.x; af[base + 1] = (__bf16)a.y;
    af[base + 2] = (__bf16)a.z; af[base + 3] = (__bf16)a.w;
    af[base + 4] = (__bf16)b.x; af[base + 5] = (__bf16)b.y;
    af[base + 6] = (__bf16)b.z; af[base + 7] = (__bf16)b.w;
}

template <bool ABF>
__global__ void k_gemm_wmma(const void* __restrict__ Asrc, const __bf16* __restrict__ Bt,
                            const float* __restrict__ bias, float* __restrict__ C,
                            int M, int N, int K, int lda, int ldc, int relu,
                            ll sA, ll sB, ll sC)
{
    const float*  A   = (const float*)Asrc  + (ll)blockIdx.z * sA;
    const __bf16* Abf = (const __bf16*)Asrc + (ll)blockIdx.z * sA;
    Bt += (ll)blockIdx.z * sB;
    C  += (ll)blockIdx.z * sC;

    const int lane = threadIdx.x;                       // 0..31
    const int row0 = (blockIdx.y * blockDim.y + threadIdx.y) * 16;
    const int col0 = blockIdx.x * 64;
    if (row0 >= M) return;
    const int rl  = lane & 15;
    const int hi  = lane >> 4;                          // 0 / 1 half-wave
    int r = row0 + rl;
    if (r >= M) r = M - 1;                              // clamp (see note above)
    int c0 = col0 + rl;          if (c0 >= N) c0 = N - 1;
    int c1 = col0 + 16 + rl;     if (c1 >= N) c1 = N - 1;
    int c2 = col0 + 32 + rl;     if (c2 >= N) c2 = N - 1;
    int c3 = col0 + 48 + rl;     if (c3 >= N) c3 = N - 1;
    const __bf16* bq0 = Bt + (ll)c0 * K;
    const __bf16* bq1 = Bt + (ll)c1 * K;
    const __bf16* bq2 = Bt + (ll)c2 * K;
    const __bf16* bq3 = Bt + (ll)c3 * K;

    v8f acc0 = {}, acc1 = {}, acc2 = {}, acc3 = {};

#pragma unroll 2
    for (int k0 = 0; k0 < K; k0 += 32) {
        // ---- A fragment 16x32 (lane<16: K 0-15 ; lane>=16: K 16-31)
        v16bf af;
        const int kh = hi * 16;
        if (ABF) {
            const __bf16* p = Abf + (ll)r * K + k0 + kh;
            v8bf lo = *(const v8bf*)p;
            v8bf hb = *(const v8bf*)(p + 8);
#pragma unroll
            for (int e = 0; e < 8; ++e) { af[e] = lo[e]; af[8 + e] = hb[e]; }
        } else {
            const float* p = A + (ll)r * lda + k0 + kh;
            __builtin_prefetch(p + 256, 0, 3);
            cvt8(af, 0, p);
            cvt8(af, 8, p + 8);
        }
        // ---- 4 B fragments 32x16 (lane<16: K k0..k0+15 ; lane>=16: +16..+31)
        const int kb = k0 + hi * 16;
        v16bf b0 = *(const v16bf*)(bq0 + kb);
        v16bf b1 = *(const v16bf*)(bq1 + kb);
        v16bf b2 = *(const v16bf*)(bq2 + kb);
        v16bf b3 = *(const v16bf*)(bq3 + kb);
        acc0 = __builtin_amdgcn_wmma_f32_16x16x32_bf16(false, af, false, b0, (short)0, acc0, false, false);
        acc1 = __builtin_amdgcn_wmma_f32_16x16x32_bf16(false, af, false, b1, (short)0, acc1, false, false);
        acc2 = __builtin_amdgcn_wmma_f32_16x16x32_bf16(false, af, false, b2, (short)0, acc2, false, false);
        acc3 = __builtin_amdgcn_wmma_f32_16x16x32_bf16(false, af, false, b3, (short)0, acc3, false, false);
    }
    // ---- store: VGPR v holds row row0+v (lanes 0-15) / row0+8+v (lanes 16-31)
#pragma unroll
    for (int t = 0; t < 4; ++t) {
        const int c = col0 + t * 16 + rl;
        if (c >= N) continue;
        const float bv = bias ? bias[c] : 0.0f;
        const v8f* ap = (t == 0) ? &acc0 : (t == 1) ? &acc1 : (t == 2) ? &acc2 : &acc3;
#pragma unroll
        for (int v = 0; v < 8; ++v) {
            const int rr = row0 + v + hi * 8;
            if (rr < M) {
                float val = (*ap)[v] + bv;
                if (relu) val = fmaxf(val, 0.0f);
                C[(ll)rr * ldc + c] = val;
            }
        }
    }
}

// ---------------------------------------------------------------------------
// utility kernels
// ---------------------------------------------------------------------------
__global__ void k_fill(float* p, float v, ll n) {
    ll i = (ll)blockIdx.x * blockDim.x + threadIdx.x;
    if (i < n) p[i] = v;
}
__global__ void k_copy(const float* a, float* b, ll n) {
    ll i = (ll)blockIdx.x * blockDim.x + threadIdx.x;
    if (i < n) b[i] = a[i];
}
__global__ void k_conv_bf16(const float* src, __bf16* dst, ll n) {
    ll i = (ll)blockIdx.x * blockDim.x + threadIdx.x;
    if (i < n) dst[i] = (__bf16)src[i];
}
// src [R][C] f32 -> dst [C][R] bf16 (batched via grid.z)
__global__ void k_transpose_bf16(const float* __restrict__ src, __bf16* __restrict__ dst,
                                 int R, int Cc, ll sSrc, ll sDst) {
    src += (ll)blockIdx.z * sSrc;
    dst += (ll)blockIdx.z * sDst;
    __shared__ float tile[16][17];
    int c = blockIdx.x * 16 + threadIdx.x;
    int r = blockIdx.y * 16 + threadIdx.y;
    if (r < R && c < Cc) tile[threadIdx.y][threadIdx.x] = src[(ll)r * Cc + c];
    __syncthreads();
    int rr = blockIdx.y * 16 + threadIdx.x;
    int cc = blockIdx.x * 16 + threadIdx.y;
    if (cc < Cc && rr < R) dst[(ll)cc * R + rr] = (__bf16)tile[threadIdx.x][threadIdx.y];
}

// asrc/adst[row,h] = sum_c h[row, h*Fh + c] * a_{src,dst}[h,c]
__global__ void k_head_dots(const float* __restrict__ h, const float* __restrict__ a_src,
                            const float* __restrict__ a_dst, float* __restrict__ asrc,
                            float* __restrict__ adst, int rows, int Fh) {
    int i = blockIdx.x * blockDim.x + threadIdx.x;
    if (i >= rows * 2) return;
    int row = i >> 1, hh = i & 1;
    const float* hp = h + (ll)row * (2 * Fh) + hh * Fh;
    const float* as = a_src + hh * Fh;
    const float* ad = a_dst + hh * Fh;
    float sa = 0.f, sd = 0.f;
    for (int c = 0; c < Fh; ++c) { float v = hp[c]; sa += v * as[c]; sd += v * ad[c]; }
    asrc[i] = sa; adst[i] = sd;
}

static __device__ __forceinline__ void atomicMaxF(float* addr, float v) {
    if (v >= 0.f) atomicMax((int*)addr, __float_as_int(v));
    else          atomicMin((unsigned int*)addr, (unsigned int)__float_as_int(v));
}
static __device__ __forceinline__ void edge_pair(const int* ei, int b, int e, int& src, int& dst) {
    if (e < E_) { src = ei[(ll)b * 2 * E_ + e]; dst = ei[(ll)b * 2 * E_ + E_ + e]; }
    else        { src = e - E_; dst = src; }
}

// e = leaky_relu(asrc[src]+adst[dst]); segment max over dst
__global__ void k_edge_logits(const int* __restrict__ ei, const float* __restrict__ asrc,
                              const float* __restrict__ adst, float* __restrict__ ebuf,
                              float* __restrict__ mbuf) {
    ll i = (ll)blockIdx.x * blockDim.x + threadIdx.x;
    if (i >= (ll)B_ * EN_ * 2) return;
    int hh = (int)(i & 1);
    ll  eh = i >> 1;
    int b = (int)(eh / EN_), e = (int)(eh % EN_);
    int src, dst; edge_pair(ei, b, e, src, dst);
    float x = asrc[((ll)b * N_ + src) * 2 + hh] + adst[((ll)b * N_ + dst) * 2 + hh];
    x = x > 0.f ? x : 0.2f * x;
    ebuf[i] = x;
    atomicMaxF(&mbuf[((ll)b * N_ + dst) * 2 + hh], x);
}
// ex = exp(e - m[dst]); segment sum into z
__global__ void k_edge_exp(const int* __restrict__ ei, float* __restrict__ ebuf,
                           const float* __restrict__ mbuf, float* __restrict__ z) {
    ll i = (ll)blockIdx.x * blockDim.x + threadIdx.x;
    if (i >= (ll)B_ * EN_ * 2) return;
    int hh = (int)(i & 1);
    ll  eh = i >> 1;
    int b = (int)(eh / EN_), e = (int)(eh % EN_);
    int src, dst; edge_pair(ei, b, e, src, dst);
    float ex = __expf(ebuf[i] - mbuf[((ll)b * N_ + dst) * 2 + hh]);
    ebuf[i] = ex;
    atomicAdd(&z[((ll)b * N_ + dst) * 2 + hh], ex);
}
// x1[dst,c] += mean_h alpha_h * h1[src, h*256 + c]    (one block per edge)
__global__ void k_edge_aggr(const int* __restrict__ ei, const float* __restrict__ ex,
                            const float* __restrict__ z, const float* __restrict__ h1,
                            float* __restrict__ x1) {
    ll be = blockIdx.x;
    int b = (int)(be / EN_), e = (int)(be % EN_);
    int src, dst; edge_pair(ei, b, e, src, dst);
    float a0 = ex[be * 2 + 0] / (z[((ll)b * N_ + dst) * 2 + 0] + 1e-16f);
    float a1 = ex[be * 2 + 1] / (z[((ll)b * N_ + dst) * 2 + 1] + 1e-16f);
    int c = threadIdx.x;  // 256
    const float* hp = h1 + ((ll)b * N_ + src) * (2 * F1_);
    float v = 0.5f * (a0 * hp[c] + a1 * hp[F1_ + c]);
    atomicAdd(&x1[((ll)b * N_ + dst) * F1_ + c], v);
}
__global__ void k_bias_relu(float* x, const float* bias, ll n, int C) {
    ll i = (ll)blockIdx.x * blockDim.x + threadIdx.x;
    if (i < n) x[i] = fmaxf(x[i] + bias[i % C], 0.0f);
}
// num[b] += w_e * (s[src].s[dst]);  dflat[src] += w_e
__global__ void k_edge_trace(const int* __restrict__ ei, const float* __restrict__ ex,
                             const float* __restrict__ z, const float* __restrict__ s1,
                             float* __restrict__ num, float* __restrict__ dflat) {
    ll i = (ll)blockIdx.x * blockDim.x + threadIdx.x;
    if (i >= (ll)B_ * EN_) return;
    int b = (int)(i / EN_), e = (int)(i % EN_);
    int src, dst; edge_pair(ei, b, e, src, dst);
    float a0 = ex[i * 2 + 0] / (z[((ll)b * N_ + dst) * 2 + 0] + 1e-16f);
    float a1 = ex[i * 2 + 1] / (z[((ll)b * N_ + dst) * 2 + 1] + 1e-16f);
    float w = 0.5f * (a0 + a1);
    const float* ps = s1 + ((ll)b * N_ + src) * K1_;
    const float* pd = s1 + ((ll)b * N_ + dst) * K1_;
    float dot = 0.f;
    for (int k = 0; k < K1_; ++k) dot += ps[k] * pd[k];
    atomicAdd(&num[b], w * dot);
    atomicAdd(&dflat[(ll)b * N_ + src], w);
}
// row softmax (L<=128), optional sum-of-squares output per row
__global__ void k_softmax_rows(float* __restrict__ x, int L, float* __restrict__ ssq) {
    ll row = blockIdx.x;
    float* p = x + row * L;
    __shared__ float sh[128];
    int t = threadIdx.x;
    float v = (t < L) ? p[t] : -3.4e38f;
    sh[t] = v; __syncthreads();
    for (int s = 64; s > 0; s >>= 1) { if (t < s) sh[t] = fmaxf(sh[t], sh[t + s]); __syncthreads(); }
    float mx = sh[0]; __syncthreads();
    float ev = (t < L) ? __expf(v - mx) : 0.f;
    sh[t] = ev; __syncthreads();
    for (int s = 64; s > 0; s >>= 1) { if (t < s) sh[t] += sh[t + s]; __syncthreads(); }
    float sum = sh[0]; __syncthreads();
    float pv = ev / sum;
    if (t < L) p[t] = pv;
    if (ssq) {
        sh[t] = pv * pv; __syncthreads();
        for (int s = 64; s > 0; s >>= 1) { if (t < s) sh[t] += sh[t + s]; __syncthreads(); }
        if (t == 0) ssq[row] = sh[0];
    }
}
// out[b] = sum_i a[b*n+i]*c[b*n+i]
__global__ void k_dot_reduce(const float* a, const float* c, float* out, int n) {
    int b = blockIdx.x, t = threadIdx.x;
    const float* pa = a + (ll)b * n;
    const float* pc = c + (ll)b * n;
    __shared__ float sh[256];
    float s = 0.f;
    for (int i = t; i < n; i += 256) s += pa[i] * pc[i];
    sh[t] = s; __syncthreads();
    for (int k = 128; k > 0; k >>= 1) { if (t < k) sh[t] += sh[t + k]; __syncthreads(); }
    if (t == 0) out[b] = sh[0];
}
// ortho loss per batch: || ss/||ss||_F - I/sqrt(k) ||_F
__global__ void k_ortho(const float* __restrict__ ss, float* __restrict__ out, int k) {
    int b = blockIdx.x, t = threadIdx.x, n = k * k;
    const float* p = ss + (ll)b * n;
    __shared__ float sh[256];
    float s = 0.f;
    for (int i = t; i < n; i += 256) { float v = p[i]; s += v * v; }
    sh[t] = s; __syncthreads();
    for (int q = 128; q > 0; q >>= 1) { if (t < q) sh[t] += sh[t + q]; __syncthreads(); }
    float ssn = sqrtf(sh[0]); __syncthreads();
    float rk = rsqrtf((float)k);
    float s2 = 0.f;
    for (int i = t; i < n; i += 256) {
        float v = p[i] / ssn;
        if (i / k == i % k) v -= rk;
        s2 += v * v;
    }
    sh[t] = s2; __syncthreads();
    for (int q = 128; q > 0; q >>= 1) { if (t < q) sh[t] += sh[t + q]; __syncthreads(); }
    if (t == 0) out[b] = sqrtf(sh[0]);
}
// BatchNorm1d over [B,C,L] (train-mode batch stats, biased var); in-place OK
__global__ void k_bn3(const float* __restrict__ x, float* __restrict__ y,
                      const float* g, const float* bb, int Bn, int C, int L, int relu) {
    int c = blockIdx.x, t = threadIdx.x;
    __shared__ float s1[256], s2[256];
    float a = 0.f, q = 0.f;
    int tot = Bn * L;
    for (int i = t; i < tot; i += 256) {
        int b = i / L, l = i % L;
        float v = x[((ll)b * C + c) * L + l];
        a += v; q += v * v;
    }
    s1[t] = a; s2[t] = q; __syncthreads();
    for (int k = 128; k > 0; k >>= 1) { if (t < k) { s1[t] += s1[t + k]; s2[t] += s2[t + k]; } __syncthreads(); }
    float mu = s1[0] / tot;
    float var = s2[0] / tot - mu * mu;
    float rs = rsqrtf(var + 1e-5f);
    float gg = g[c], bv = bb[c];
    for (int i = t; i < tot; i += 256) {
        int b = i / L, l = i % L;
        ll idx = ((ll)b * C + c) * L + l;
        float v = (x[idx] - mu) * rs * gg + bv;
        if (relu) v = fmaxf(v, 0.f);
        y[idx] = v;
    }
}
// dense GAT over 100 pooled nodes; writes out [B,100,128] and adj2[src,dst]
__global__ void k_gat_dense(const float* __restrict__ h, const float* __restrict__ asrc,
                            const float* __restrict__ adst, const float* __restrict__ bias,
                            float* __restrict__ out, float* __restrict__ adj) {
    const int d = blockIdx.x, b = blockIdx.y, t = threadIdx.x;  // block 128
    __shared__ float al0[128], al1[128], red[128];
    const ll rb = (ll)b * K1_;
    float e0 = -3.4e38f, e1 = -3.4e38f;
    if (t < K1_) {
        float x0 = adst[(rb + d) * 2 + 0] + asrc[(rb + t) * 2 + 0];
        float x1 = adst[(rb + d) * 2 + 1] + asrc[(rb + t) * 2 + 1];
        e0 = x0 > 0.f ? x0 : 0.2f * x0;
        e1 = x1 > 0.f ? x1 : 0.2f * x1;
    }
    red[t] = e0; __syncthreads();
    for (int s = 64; s > 0; s >>= 1) { if (t < s) red[t] = fmaxf(red[t], red[t + s]); __syncthreads(); }
    float m0 = red[0]; __syncthreads();
    float x0e = (t < K1_) ? __expf(e0 - m0) : 0.f;
    red[t] = x0e; __syncthreads();
    for (int s = 64; s > 0; s >>= 1) { if (t < s) red[t] += red[t + s]; __syncthreads(); }
    float z0 = red[0]; __syncthreads();
    al0[t] = x0e / z0;
    red[t] = e1; __syncthreads();
    for (int s = 64; s > 0; s >>= 1) { if (t < s) red[t] = fmaxf(red[t], red[t + s]); __syncthreads(); }
    float m1 = red[0]; __syncthreads();
    float x1e = (t < K1_) ? __expf(e1 - m1) : 0.f;
    red[t] = x1e; __syncthreads();
    for (int s = 64; s > 0; s >>= 1) { if (t < s) red[t] += red[t + s]; __syncthreads(); }
    float z1 = red[0]; __syncthreads();
    al1[t] = x1e / z1;
    __syncthreads();
    if (t < K1_) adj[(rb + t) * K1_ + d] = 0.5f * (al0[t] + al1[t]);
    float a0 = 0.f, a1 = 0.f;
    for (int s = 0; s < K1_; ++s) {
        const float* hp = h + (rb + s) * (2 * F2_);
        a0 += al0[s] * hp[t];
        a1 += al1[s] * hp[F2_ + t];
    }
    out[(rb + d) * F2_ + t] = 0.5f * (a0 + a1) + bias[t];
}
// pool-2 direct kernels
__global__ void k_xp2(const float* __restrict__ s2, const float* __restrict__ x2,
                      float* __restrict__ xp2) {
    int i = blockIdx.x * blockDim.x + threadIdx.x;
    if (i >= B_ * K2_ * F2_) return;
    int b = i / (K2_ * F2_), r = i % (K2_ * F2_), k = r / F2_, c = r % F2_;
    float acc = 0.f;
    for (int n = 0; n < K1_; ++n)
        acc += s2[((ll)b * K1_ + n) * K2_ + k] * x2[((ll)b * K1_ + n) * F2_ + c];
    xp2[i] = acc;
}
__global__ void k_dflat2(const float* __restrict__ adj, float* __restrict__ dflat) {
    int i = blockIdx.x * blockDim.x + threadIdx.x;
    if (i >= B_ * K1_) return;
    float s = 0.f;
    for (int j = 0; j < K1_; ++j) s += adj[(ll)i * K1_ + j];
    dflat[i] = s;
}
__global__ void k_num2(const float* __restrict__ adj, const float* __restrict__ s2,
                       float* __restrict__ num) {
    int i = blockIdx.x * blockDim.x + threadIdx.x;
    if (i >= B_ * K1_ * K1_) return;
    int b = i / (K1_ * K1_), r = i % (K1_ * K1_), ii = r / K1_, j = r % K1_;
    float dot = 0.f;
    for (int k = 0; k < K2_; ++k)
        dot += s2[((ll)b * K1_ + ii) * K2_ + k] * s2[((ll)b * K1_ + j) * K2_ + k];
    atomicAdd(&num[b], adj[i] * dot);
}
__global__ void k_ss2(const float* __restrict__ s2, float* __restrict__ ss) {
    int i = blockIdx.x * blockDim.x + threadIdx.x;
    if (i >= B_ * K2_ * K2_) return;
    int b = i / (K2_ * K2_), r = i % (K2_ * K2_), k = r / K2_, j = r % K2_;
    float acc = 0.f;
    for (int n = 0; n < K1_; ++n)
        acc += s2[((ll)b * K1_ + n) * K2_ + k] * s2[((ll)b * K1_ + n) * K2_ + j];
    ss[i] = acc;
}
__global__ void k_build_xt(const float* __restrict__ cls, const float* __restrict__ xp2,
                           float* __restrict__ xt) {
    int i = blockIdx.x * blockDim.x + threadIdx.x;
    if (i >= B_ * S_ * D_) return;
    int b = i / (S_ * D_), r = i % (S_ * D_), s = r / D_, c = r % D_;
    xt[i] = (s == 0) ? cls[c] : xp2[((ll)b * K2_ + (s - 1)) * D_ + c];
}
// per-(batch,head) attention, one thread per query row
__global__ void k_attn(const float* __restrict__ q, int qs, const float* __restrict__ kk,
                       int ks, const float* __restrict__ vv, int vs,
                       float* __restrict__ out, int Sq, int Skv) {
    int b = blockIdx.x, hh = blockIdx.y, t = threadIdx.x;
    if (t >= Sq) return;
    const float* qp = q  + ((ll)b * Sq + t) * qs + hh * 64;
    const float* kb = kk + (ll)b * Skv * ks + hh * 64;
    const float* vb = vv + (ll)b * Skv * vs + hh * 64;
    float sc[64];
    float mx = -3.4e38f;
    for (int j = 0; j < Skv; ++j) {
        const float* kp = kb + (ll)j * ks;
        float s = 0.f;
        for (int c = 0; c < 64; ++c) s += qp[c] * kp[c];
        s *= 0.125f;                 // 1/sqrt(64)
        sc[j] = s;
        mx = fmaxf(mx, s);
    }
    float sum = 0.f;
    for (int j = 0; j < Skv; ++j) { sc[j] = __expf(sc[j] - mx); sum += sc[j]; }
    float inv = 1.f / sum;
    float* op = out + ((ll)b * Sq + t) * D_ + hh * 64;
    for (int c = 0; c < 64; ++c) {
        float a = 0.f;
        for (int j = 0; j < Skv; ++j) a += sc[j] * vb[(ll)j * vs + c];
        op[c] = a * inv;
    }
}
// x = LayerNorm(x + addv)   (rows of 128; addv may be null)
__global__ void k_add_ln(float* __restrict__ x, const float* __restrict__ addv,
                         const float* __restrict__ g, const float* __restrict__ bb) {
    ll row = blockIdx.x;
    int t = threadIdx.x;  // 128
    ll idx = row * D_ + t;
    float v = x[idx] + (addv ? addv[idx] : 0.f);
    __shared__ float sh[128];
    sh[t] = v; __syncthreads();
    for (int s = 64; s > 0; s >>= 1) { if (t < s) sh[t] += sh[t + s]; __syncthreads(); }
    float mu = sh[0] / D_; __syncthreads();
    float dv = v - mu;
    sh[t] = dv * dv; __syncthreads();
    for (int s = 64; s > 0; s >>= 1) { if (t < s) sh[t] += sh[t + s]; __syncthreads(); }
    float var = sh[0] / D_;
    x[idx] = dv * rsqrtf(var + 1e-5f) * g[t] + bb[t];
}
// batch_norm2 over 8 cls tokens
__global__ void k_bn2(const float* __restrict__ y, const float* g, const float* bb,
                      float* __restrict__ feat) {
    int c = threadIdx.x;  // 128
    float s = 0.f, q = 0.f;
    for (int b = 0; b < B_; ++b) { float v = y[(ll)b * S_ * D_ + c]; s += v; q += v * v; }
    float mu = s / B_, var = q / B_ - mu * mu, rs = rsqrtf(var + 1e-5f);
    for (int b = 0; b < B_; ++b) {
        float v = y[(ll)b * S_ * D_ + c];
        feat[b * D_ + c] = (v - mu) * rs * g[c] + bb[c];
    }
}
__global__ void k_pred(const float* feat, const float* fcw, const float* fcb, float* out) {
    int b = threadIdx.x;
    if (b >= B_) return;
    float s = fcb[0];
    for (int c = 0; c < D_; ++c) s += feat[b * D_ + c] * fcw[c];
    out[b] = 1.f / (1.f + __expf(-s));
}
__global__ void k_losses(const float* num1, const float* den1, const float* o1,
                         const float* num2, const float* den2, const float* o2,
                         float* outp) {
    if (threadIdx.x != 0 || blockIdx.x != 0) return;
    float mc = 0.f, oo = 0.f;
    for (int b = 0; b < B_; ++b) {
        mc += -num1[b] / den1[b] - num2[b] / den2[b];
        oo += o1[b] + o2[b];
    }
    outp[8] = mc / B_;
    outp[9] = oo / B_;
}

// ---------------------------------------------------------------------------
extern "C" void kernel_launch(void* const* d_in, const int* in_sizes, int n_in,
                              void* d_out, int out_size, void* d_ws, size_t ws_size,
                              hipStream_t stream) {
    (void)in_sizes; (void)n_in; (void)out_size; (void)ws_size;
    auto F = [&](int i) { return (const float*)d_in[i]; };
    const float* nodes = F(0);
    const int*   ei    = (const int*)d_in[1];
    float* outp = (float*)d_out;

    // ---- workspace bump allocator (256B aligned) ----
    char* wsp = (char*)d_ws;
    size_t off = 0;
    auto alignup = [&]() { off = (off + 255) & ~(size_t)255; };
    auto allocF = [&](size_t n) -> float*  { float*  p = (float*)(wsp + off);  off += n * 4; alignup(); return p; };
    auto allocH = [&](size_t n) -> __bf16* { __bf16* p = (__bf16*)(wsp + off); off += n * 2; alignup(); return p; };

    // bf16 weight buffers
    __bf16* W1t    = allocH((size_t)FIN_ * (2 * F1_));   // [512][512]
    __bf16* mlp1wt = allocH((size_t)K1_ * F1_);          // [100][256]
    __bf16* W2t    = allocH((size_t)F1_ * (2 * F2_));    // [256][256]
    __bf16* mlp2wt = allocH((size_t)K2_ * F2_);          // [50][128]
    __bf16 *encQ[2], *encO[2], *encW1[2], *encW2[2];
    __bf16 *decSQ[2], *decSO[2], *decCQ[2], *decCO[2], *decW1[2], *decW2[2];
    for (int l = 0; l < 2; ++l) {
        encQ[l] = allocH(384 * D_); encO[l] = allocH(D_ * D_);
        encW1[l] = allocH(D_ * D_); encW2[l] = allocH(D_ * D_);
    }
    for (int l = 0; l < 2; ++l) {
        decSQ[l] = allocH(384 * D_); decSO[l] = allocH(D_ * D_);
        decCQ[l] = allocH(384 * D_); decCO[l] = allocH(D_ * D_);
        decW1[l] = allocH(D_ * D_);  decW2[l] = allocH(D_ * D_);
    }
    // f32 activations
    float* h1    = allocF((size_t)B_ * N_ * (2 * F1_));
    float* asrc1 = allocF((size_t)B_ * N_ * 2);
    float* adst1 = allocF((size_t)B_ * N_ * 2);
    float* m1    = allocF((size_t)B_ * N_ * 2);
    float* z1    = allocF((size_t)B_ * N_ * 2);
    float* ex1   = allocF((size_t)B_ * EN_ * 2);
    float* x1    = allocF((size_t)B_ * N_ * F1_);
    float* s1    = allocF((size_t)B_ * N_ * K1_);
    float* ssq1  = allocF((size_t)B_ * N_);
    float* dflat1= allocF((size_t)B_ * N_);
    __bf16* x1t  = allocH((size_t)B_ * F1_ * N_);
    __bf16* s1t  = allocH((size_t)B_ * K1_ * N_);
    float* ss1   = allocF((size_t)B_ * K1_ * K1_);
    float* xp1   = allocF((size_t)B_ * K1_ * F1_);
    float* h2raw = allocF((size_t)B_ * K1_ * (2 * F2_));
    float* asrc2 = allocF((size_t)B_ * K1_ * 2);
    float* adst2 = allocF((size_t)B_ * K1_ * 2);
    float* h2    = allocF((size_t)B_ * K1_ * F2_);
    float* x2    = allocF((size_t)B_ * K1_ * F2_);
    float* adj2  = allocF((size_t)B_ * K1_ * K1_);
    float* s2    = allocF((size_t)B_ * K1_ * K2_);
    float* ssq2  = allocF((size_t)B_ * K1_);
    float* dflat2= allocF((size_t)B_ * K1_);
    float* ss2   = allocF((size_t)B_ * K2_ * K2_);
    float* xp2   = allocF((size_t)B_ * K2_ * F2_);
    float* xt    = allocF((size_t)B_ * S_ * D_);
    float* mbuf  = allocF((size_t)B_ * S_ * D_);
    float* ybuf  = allocF((size_t)B_ * S_ * D_);
    float* tq    = allocF((size_t)B_ * S_ * 384);
    float* t2    = allocF((size_t)B_ * S_ * D_);
    float* t3    = allocF((size_t)B_ * S_ * D_);
    float* num1  = allocF(B_); float* den1 = allocF(B_); float* o1a = allocF(B_);
    float* num2  = allocF(B_); float* den2 = allocF(B_); float* o2a = allocF(B_);
    float* feat  = allocF(B_ * D_);

    // gemm: A is f32 row-major
    auto gemm = [&](const float* A, const __bf16* Bt, const float* bias,
                    float* C, int M, int Nc, int K, int lda, int ldc, int relu,
                    ll sA, ll sB, ll sC, int batch) {
        dim3 blk(32, 4, 1), grd((Nc + 63) / 64, (M + 63) / 64, batch);
        k_gemm_wmma<false><<<grd, blk, 0, stream>>>((const void*)A, Bt, bias, C, M, Nc, K,
                                                    lda, ldc, relu, sA, sB, sC);
    };
    // gemmB: A is bf16 row-major [M][K]
    auto gemmB = [&](const __bf16* A, const __bf16* Bt, const float* bias,
                     float* C, int M, int Nc, int K, int ldc, int relu,
                     ll sA, ll sB, ll sC, int batch) {
        dim3 blk(32, 4, 1), grd((Nc + 63) / 64, (M + 63) / 64, batch);
        k_gemm_wmma<true><<<grd, blk, 0, stream>>>((const void*)A, Bt, bias, C, M, Nc, K,
                                                   0, ldc, relu, sA, sB, sC);
    };
    auto fill = [&](float* p, float v, ll n) {
        k_fill<<<(int)((n + 255) / 256), 256, 0, stream>>>(p, v, n);
    };
    auto conv = [&](const float* s, __bf16* d, ll n) {
        k_conv_bf16<<<(int)((n + 255) / 256), 256, 0, stream>>>(s, d, n);
    };
    auto tr = [&](const float* s, __bf16* d, int R, int C, ll sS, ll sD, int batch) {
        dim3 blk(16, 16, 1), grd((C + 15) / 16, (R + 15) / 16, batch);
        k_transpose_bf16<<<grd, blk, 0, stream>>>(s, d, R, C, sS, sD);
    };

    // ---- weight prep ----
    tr(F(2),  W1t,    FIN_, 2 * F1_, 0, 0, 1);             // W1^T
    tr(F(10), mlp1wt, F1_,  K1_,     0, 0, 1);
    tr(F(6),  W2t,    F1_,  2 * F2_, 0, 0, 1);
    tr(F(12), mlp2wt, F2_,  K2_,     0, 0, 1);
    for (int l = 0; l < 2; ++l) {
        int p0 = 21 + l * 12;
        conv(F(p0 + 0), encQ[l], 384 * D_);
        conv(F(p0 + 2), encO[l], D_ * D_);
        tr(F(p0 + 4), encW1[l], D_, D_, 0, 0, 1);
        tr(F(p0 + 6), encW2[l], D_, D_, 0, 0, 1);
    }
    for (int l = 0; l < 2; ++l) {
        int p0 = 45 + l * 18;
        conv(F(p0 + 0), decSQ[l], 384 * D_);
        conv(F(p0 + 2), decSO[l], D_ * D_);
        conv(F(p0 + 4), decCQ[l], 384 * D_);
        conv(F(p0 + 6), decCO[l], D_ * D_);
        tr(F(p0 + 8),  decW1[l], D_, D_, 0, 0, 1);
        tr(F(p0 + 10), decW2[l], D_, D_, 0, 0, 1);
    }

    // ---- phase 1: sparse GAT ----
    gemm(nodes, W1t, nullptr, h1, B_ * N_, 2 * F1_, FIN_, FIN_, 2 * F1_, 0, 0, 0, 0, 1);
    k_head_dots<<<(B_ * N_ * 2 + 255) / 256, 256, 0, stream>>>(h1, F(3), F(4), asrc1, adst1,
                                                               B_ * N_, F1_);
    fill(m1, -3.4e38f, (ll)B_ * N_ * 2);
    fill(z1, 0.f, (ll)B_ * N_ * 2);
    fill(x1, 0.f, (ll)B_ * N_ * F1_);
    fill(num1, 0.f, B_); fill(num2, 0.f, B_);
    fill(dflat1, 0.f, (ll)B_ * N_);
    {
        ll tot = (ll)B_ * EN_ * 2;
        k_edge_logits<<<(int)((tot + 255) / 256), 256, 0, stream>>>(ei, asrc1, adst1, ex1, m1);
        k_edge_exp<<<(int)((tot + 255) / 256), 256, 0, stream>>>(ei, ex1, m1, z1);
    }
    k_edge_aggr<<<B_ * EN_, 256, 0, stream>>>(ei, ex1, z1, h1, x1);
    k_bias_relu<<<(int)(((ll)B_ * N_ * F1_ + 255) / 256), 256, 0, stream>>>(x1, F(5),
                                                                            (ll)B_ * N_ * F1_, F1_);
    // cluster assignments + pooling statistics
    gemm(x1, mlp1wt, F(11), s1, B_ * N_, K1_, F1_, F1_, K1_, 0, 0, 0, 0, 1);
    k_softmax_rows<<<B_ * N_, 128, 0, stream>>>(s1, K1_, ssq1);
    k_edge_trace<<<(B_ * EN_ + 255) / 256, 256, 0, stream>>>(ei, ex1, z1, s1, num1, dflat1);
    k_dot_reduce<<<B_, 256, 0, stream>>>(dflat1, ssq1, den1, N_);
    tr(s1, s1t, N_, K1_, (ll)N_ * K1_, (ll)K1_ * N_, B_);
    tr(x1, x1t, N_, F1_, (ll)N_ * F1_, (ll)F1_ * N_, B_);
    gemmB(s1t, s1t, nullptr, ss1, K1_, K1_, N_, K1_, 0,
          (ll)K1_ * N_, (ll)K1_ * N_, (ll)K1_ * K1_, B_);
    k_ortho<<<B_, 256, 0, stream>>>(ss1, o1a, K1_);
    gemmB(s1t, x1t, nullptr, xp1, K1_, F1_, N_, F1_, 0,
          (ll)K1_ * N_, (ll)F1_ * N_, (ll)K1_ * F1_, B_);
    k_bn3<<<K1_, 256, 0, stream>>>(xp1, xp1, F(14), F(15), B_, K1_, F1_, 0);

    // ---- phase 2: dense GAT on pooled graph ----
    gemm(xp1, W2t, nullptr, h2raw, B_ * K1_, 2 * F2_, F1_, F1_, 2 * F2_, 0, 0, 0, 0, 1);
    k_head_dots<<<(B_ * K1_ * 2 + 255) / 256, 256, 0, stream>>>(h2raw, F(7), F(8), asrc2, adst2,
                                                                B_ * K1_, F2_);
    k_gat_dense<<<dim3(K1_, B_), 128, 0, stream>>>(h2raw, asrc2, adst2, F(9), h2, adj2);
    k_bn3<<<K1_, 256, 0, stream>>>(h2, x2, F(14), F(15), B_, K1_, F2_, 1);
    gemm(x2, mlp2wt, F(13), s2, B_ * K1_, K2_, F2_, F2_, K2_, 0, 0, 0, 0, 1);
    k_softmax_rows<<<B_ * K1_, 128, 0, stream>>>(s2, K2_, ssq2);
    k_xp2<<<(B_ * K2_ * F2_ + 255) / 256, 256, 0, stream>>>(s2, x2, xp2);
    k_dflat2<<<(B_ * K1_ + 255) / 256, 256, 0, stream>>>(adj2, dflat2);
    k_num2<<<(B_ * K1_ * K1_ + 255) / 256, 256, 0, stream>>>(adj2, s2, num2);
    k_dot_reduce<<<B_, 256, 0, stream>>>(dflat2, ssq2, den2, K1_);
    k_ss2<<<(B_ * K2_ * K2_ + 255) / 256, 256, 0, stream>>>(s2, ss2);
    k_ortho<<<B_, 256, 0, stream>>>(ss2, o2a, K2_);

    // ---- phase 3: transformer ----
    const int BS = B_ * S_;
    const ll  nXT = (ll)B_ * S_ * D_;
    k_build_xt<<<(int)((nXT + 255) / 256), 256, 0, stream>>>(F(18), xp2, xt);
    k_copy<<<(int)((nXT + 255) / 256), 256, 0, stream>>>(xt, mbuf, nXT);
    for (int l = 0; l < 2; ++l) {
        int p0 = 21 + l * 12;
        gemm(mbuf, encQ[l], F(p0 + 1), tq, BS, 384, D_, D_, 384, 0, 0, 0, 0, 1);
        k_attn<<<dim3(B_, 2), 64, 0, stream>>>(tq, 384, tq + 128, 384, tq + 256, 384, t2, S_, S_);
        gemm(t2, encO[l], F(p0 + 3), t3, BS, D_, D_, D_, D_, 0, 0, 0, 0, 1);
        k_add_ln<<<BS, 128, 0, stream>>>(mbuf, t3, F(p0 + 8), F(p0 + 9));
        gemm(mbuf, encW1[l], F(p0 + 5), t2, BS, D_, D_, D_, D_, 1, 0, 0, 0, 1);
        gemm(t2, encW2[l], F(p0 + 7), t3, BS, D_, D_, D_, D_, 0, 0, 0, 0, 1);
        k_add_ln<<<BS, 128, 0, stream>>>(mbuf, t3, F(p0 + 10), F(p0 + 11));
    }
    k_add_ln<<<BS, 128, 0, stream>>>(mbuf, nullptr, F(81), F(82));
    k_copy<<<(int)((nXT + 255) / 256), 256, 0, stream>>>(xt, ybuf, nXT);
    for (int l = 0; l < 2; ++l) {
        int p0 = 45 + l * 18;
        // self-attention
        gemm(ybuf, decSQ[l], F(p0 + 1), tq, BS, 384, D_, D_, 384, 0, 0, 0, 0, 1);
        k_attn<<<dim3(B_, 2), 64, 0, stream>>>(tq, 384, tq + 128, 384, tq + 256, 384, t2, S_, S_);
        gemm(t2, decSO[l], F(p0 + 3), t3, BS, D_, D_, D_, D_, 0, 0, 0, 0, 1);
        k_add_ln<<<BS, 128, 0, stream>>>(ybuf, t3, F(p0 + 12), F(p0 + 13));
        // cross-attention (q from y, kv from encoder memory)
        gemm(ybuf, decCQ[l], F(p0 + 5), tq, BS, 128, D_, D_, 384, 0, 0, 0, 0, 1);
        gemm(mbuf, decCQ[l] + 128 * D_, F(p0 + 5) + 128, tq + 128, BS, 256, D_, D_, 384,
             0, 0, 0, 0, 1);
        k_attn<<<dim3(B_, 2), 64, 0, stream>>>(tq, 384, tq + 128, 384, tq + 256, 384, t2, S_, S_);
        gemm(t2, decCO[l], F(p0 + 7), t3, BS, D_, D_, D_, D_, 0, 0, 0, 0, 1);
        k_add_ln<<<BS, 128, 0, stream>>>(ybuf, t3, F(p0 + 14), F(p0 + 15));
        // ffn
        gemm(ybuf, decW1[l], F(p0 + 9), t2, BS, D_, D_, D_, D_, 1, 0, 0, 0, 1);
        gemm(t2, decW2[l], F(p0 + 11), t3, BS, D_, D_, D_, D_, 0, 0, 0, 0, 1);
        k_add_ln<<<BS, 128, 0, stream>>>(ybuf, t3, F(p0 + 16), F(p0 + 17));
    }
    k_add_ln<<<BS, 128, 0, stream>>>(ybuf, nullptr, F(83), F(84));

    // ---- head + losses ----
    k_bn2<<<1, 128, 0, stream>>>(ybuf, F(16), F(17), feat);
    k_pred<<<1, 32, 0, stream>>>(feat, F(19), F(20), outp);
    k_losses<<<1, 1, 0, stream>>>(num1, den1, o1a, num2, den2, o2a, outp);
}
